// MultiHeadSelfAttention_79834852098682
// MI455X (gfx1250) — matrix operand order repaired and checked by reading, not compiled, not run on hardware
//
#include <hip/hip_runtime.h>
#include <hip/hip_bf16.h>

typedef __bf16 bf16_t;
typedef __attribute__((ext_vector_type(16))) __bf16 v16bf;
typedef __attribute__((ext_vector_type(8)))  float   v8f;
typedef __attribute__((ext_vector_type(4)))  __bf16 bf16x4;
typedef __attribute__((ext_vector_type(4)))  unsigned uint32x4;
typedef __attribute__((ext_vector_type(8)))  int      int32x8;
typedef __attribute__((ext_vector_type(4)))  int      int32x4;

#define S_LEN   2048
#define D_MODEL 1024
#define N_HEADS 16
#define D_HEAD  64

#if __has_builtin(__builtin_amdgcn_tensor_load_to_lds) && \
    __has_builtin(__builtin_amdgcn_s_wait_tensorcnt)
#define HAVE_TDM 1
#else
#define HAVE_TDM 0
#endif

union FragU { v16bf v; unsigned u[8]; };

// A-fragment (16x32 bf16, MxK) from LDS, row stride `rs` elements.
// ISA layout: lanes 0-15 = rows, VGPR i holds K pair; K = (i<4 ? 2i : 16+2(i-4)) + 8*(lane>>4)
__device__ __forceinline__ v16bf lds_load_a_frag(const bf16_t* row_base, int rs, int kbase, int lane) {
    const int m = lane & 15, hf = lane >> 4;
    FragU f;
    const unsigned* p = (const unsigned*)(row_base + m * rs + kbase);
#pragma unroll
    for (int i = 0; i < 8; ++i) {
        const int k = ((i < 4) ? (2 * i) : (16 + 2 * (i - 4))) + 8 * hf;
        f.u[i] = p[k >> 1];
    }
    return f.v;
}

// B-fragment (32x16 bf16, KxN) from LDS stored transposed ([n][k], stride rs).
// ISA layout: lane = col (n) in each half; elements 0..7 = K 8h..8h+7, elements 8..15 = K 16+8h..16+8h+7.
__device__ __forceinline__ v16bf lds_load_b_fragT(const bf16_t* nt_base, int rs, int kbase, int lane) {
    const int n = lane & 15, kh = lane >> 4;
    const bf16_t* p = nt_base + n * rs + kbase + kh * 8;
    const uint4 lo = *(const uint4*)(p);
    const uint4 hi = *(const uint4*)(p + 16);
    FragU f;
    f.u[0] = lo.x; f.u[1] = lo.y; f.u[2] = lo.z; f.u[3] = lo.w;
    f.u[4] = hi.x; f.u[5] = hi.y; f.u[6] = hi.z; f.u[7] = hi.w;
    return f.v;
}

__device__ __forceinline__ v8f wmma_bf16(v16bf a, v16bf b, v8f c) {
    return __builtin_amdgcn_wmma_f32_16x16x32_bf16(false, a, false, b, (short)0, c, false, false);
}

#if HAVE_TDM
// TDM 2D tile load: global (row stride gstride elems, bf16) -> LDS with +8 elem row pad
// (pad_interval code 4 = 32 dwords = 64 bf16 row; pad_amount code 3 = 4 dwords = 8 bf16),
// giving LDS row stride 72 elements to match the fragment loaders above.
// Issue from ONE wave only; follow with s_wait_tensorcnt(0) + block barrier.
__device__ __forceinline__ void tdm_load_tile_2d(const bf16_t* gsrc, bf16_t* ldst,
                                                 int tile_w, int tile_h, int gstride) {
    unsigned long long ga = (unsigned long long)(size_t)gsrc;
    uint32x4 g0;
    g0[0] = 1u;                                             // count=1, no gather
    g0[1] = (unsigned)(size_t)ldst;                         // lds_addr (bytes)
    g0[2] = (unsigned)ga;                                   // global_addr[31:0]
    g0[3] = (unsigned)((ga >> 32) & 0x01FFFFFFull) | (2u << 30); // addr[56:32] | type=2
    int32x8 g1;
    g1[0] = (1 << 16) | (1 << 20) | (4 << 22) | (3 << 25);  // data_size=2B, pad_en, 32dw/+4dw
    g1[1] = 0;                                              // tensor_dim0 = 1<<20 (low 16 = 0)
    g1[2] = (0x100000 >> 16);                               // tensor_dim0 hi | tensor_dim1 lo
    g1[3] = (0x100000 >> 16) | (tile_w << 16);              // tensor_dim1 hi | tile_dim0
    g1[4] = tile_h;                                         // tile_dim1 | tile_dim2=0
    g1[5] = gstride;                                        // tensor_dim0_stride (elems)
    g1[6] = 0;
    g1[7] = 0;
    int32x4 gz4 = {};
    int32x8 gz8 = {};
    // 6-arg form (clang-23 / therock headers): groups 2/3 + trailing v8i32 unused for 2D tiles.
    __builtin_amdgcn_tensor_load_to_lds(g0, g1, gz4, gz4, gz8, 0);
}
#endif

// ---------------------------------------------------------------- conversion
__global__ __launch_bounds__(256) void cvt_f32_bf16(const float* __restrict__ in,
                                                    bf16_t* __restrict__ out, int n4) {
    int i = blockIdx.x * blockDim.x + threadIdx.x;
    if (i < n4) {
        float4 v = ((const float4*)in)[i];
        bf16x4 o;
        o[0] = (bf16_t)v.x; o[1] = (bf16_t)v.y; o[2] = (bf16_t)v.z; o[3] = (bf16_t)v.w;
        ((bf16x4*)out)[i] = o;
    }
}

// ---------------------------------------------------------------- GEMM C = (A*B + bias)*oscale
// A: MxK bf16 row-major; B: KxN bf16 row-major; 128x128 tile per block, TK=64.
template <bool OUT_F32>
__global__ __launch_bounds__(256) void gemm_bf16(const bf16_t* __restrict__ A,
                                                 const bf16_t* __restrict__ Bw,
                                                 const float* __restrict__ bias,
                                                 void* __restrict__ Cout,
                                                 int M, int N, int K, float oscale) {
    __shared__ __align__(16) bf16_t As[128 * 72];
    __shared__ __align__(16) bf16_t Bt[128 * 72];   // transposed: [n][k]
    const int tid = threadIdx.x, lane = tid & 31, w = tid >> 5;
    const int wm = w & 3, wn = w >> 2;              // 4x2 wave grid
    const int mBase = blockIdx.y * 128, nBase = blockIdx.x * 128;

    v8f acc[2][4];
#pragma unroll
    for (int mt = 0; mt < 2; ++mt)
#pragma unroll
        for (int nt = 0; nt < 4; ++nt) acc[mt][nt] = {};

    for (int k0 = 0; k0 < K; k0 += 64) {
        __syncthreads();
#if HAVE_TDM
        if (w == 0)                                 // A tile 128x64 via Tensor Data Mover
            tdm_load_tile_2d(A + (size_t)mBase * K + k0, As, 64, 128, K);
#else
#pragma unroll
        for (int i = 0; i < 4; ++i) {               // A tile: 128x64, 16B chunks
            int c = i * 256 + tid;
            int m = c >> 3, kc = c & 7;
            *(uint4*)(As + m * 72 + kc * 8) =
                *(const uint4*)(A + (size_t)(mBase + m) * K + k0 + kc * 8);
        }
#endif
#pragma unroll
        for (int i = 0; i < 4; ++i) {               // B tile: 64x128 -> transposed scatter
            int c = i * 256 + tid;
            int kr = c >> 4, nc = c & 15;
            uint4 t = *(const uint4*)(Bw + (size_t)(k0 + kr) * N + nBase + nc * 8);
            const bf16_t* e = (const bf16_t*)&t;
#pragma unroll
            for (int j = 0; j < 8; ++j) Bt[(nc * 8 + j) * 72 + kr] = e[j];
        }
#if HAVE_TDM
        if (w == 0) __builtin_amdgcn_s_wait_tensorcnt((short)0);
#endif
        __syncthreads();
#pragma unroll
        for (int kk = 0; kk < 2; ++kk) {
            v16bf af[2], bfr[4];
#pragma unroll
            for (int mt = 0; mt < 2; ++mt)
                af[mt] = lds_load_a_frag(As + (wm * 32 + mt * 16) * 72, 72, kk * 32, lane);
#pragma unroll
            for (int nt = 0; nt < 4; ++nt)
                bfr[nt] = lds_load_b_fragT(Bt + (wn * 64 + nt * 16) * 72, 72, kk * 32, lane);
#pragma unroll
            for (int mt = 0; mt < 2; ++mt)
#pragma unroll
                for (int nt = 0; nt < 4; ++nt)
                    acc[mt][nt] = wmma_bf16(af[mt], bfr[nt], acc[mt][nt]);
        }
    }
    const int hf = lane >> 4, ln = lane & 15;
#pragma unroll
    for (int mt = 0; mt < 2; ++mt)
#pragma unroll
        for (int nt = 0; nt < 4; ++nt) {
            int col = nBase + wn * 64 + nt * 16 + ln;
            float bv = bias[col];
#pragma unroll
            for (int r = 0; r < 8; ++r) {
                int row = mBase + wm * 32 + mt * 16 + r + 8 * hf;
                float v = (acc[mt][nt][r] + bv) * oscale;
                if (OUT_F32) ((float*)Cout)[(size_t)row * N + col] = v;
                else         ((bf16_t*)Cout)[(size_t)row * N + col] = (bf16_t)v;
            }
        }
}

// ---------------------------------------------------------------- flash attention
// One block = 128 q-rows of one (b,h). 8 waves, each wave owns 16 q-rows x all 64 dv.
// Q is pre-scaled by 1/sqrt(DK) in its projection epilogue.
__global__ __launch_bounds__(256) void flash_attn(const bf16_t* __restrict__ Q,
                                                  const bf16_t* __restrict__ Km,
                                                  const bf16_t* __restrict__ V,
                                                  bf16_t* __restrict__ ctx) {
    __shared__ __align__(16) bf16_t Qs[128 * 72];
    __shared__ __align__(16) bf16_t Ks[64 * 72];    // [key][dk]
    __shared__ __align__(16) bf16_t Vt[64 * 72];    // [dv][key] (transposed)
    __shared__ __align__(16) bf16_t Ps[128 * 72];   // per-wave 16x64 P scratch
    const int tid = threadIdx.x, lane = tid & 31, w = tid >> 5;
    const int qBase = blockIdx.x * 128;
    const int b = blockIdx.y >> 4, h = blockIdx.y & 15;
    const size_t headOff = (size_t)b * S_LEN * D_MODEL + (size_t)h * D_HEAD;
    const int hf = lane >> 4, ln = lane & 15;
    const int myRowBase = qBase + w * 16 + 8 * hf;

#if HAVE_TDM
    if (w == 0) {                                   // stage Q tile 128x64 via TDM
        tdm_load_tile_2d(Q + headOff + (size_t)qBase * D_MODEL, Qs, 64, 128, D_MODEL);
        __builtin_amdgcn_s_wait_tensorcnt((short)0);
    }
#else
#pragma unroll
    for (int i = 0; i < 4; ++i) {                   // stage Q tile 128x64
        int c = i * 256 + tid;
        int m = c >> 3, kc = c & 7;
        *(uint4*)(Qs + m * 72 + kc * 8) =
            *(const uint4*)(Q + headOff + (size_t)(qBase + m) * D_MODEL + kc * 8);
    }
#endif
    __syncthreads();
    v16bf aq[2];
    aq[0] = lds_load_a_frag(Qs + w * 16 * 72, 72, 0, lane);
    aq[1] = lds_load_a_frag(Qs + w * 16 * 72, 72, 32, lane);

    v8f o[4];
#pragma unroll
    for (int nt = 0; nt < 4; ++nt) o[nt] = {};
    float rmax[8], rsum[8];
#pragma unroll
    for (int r = 0; r < 8; ++r) { rmax[r] = -3.0e38f; rsum[r] = 0.f; }

    for (int kb = 0; kb < qBase + 128; kb += 64) {  // causal: key tiles up to diagonal
        __syncthreads();
#if HAVE_TDM
        if (w == 0)                                 // K tile 64x64 via TDM
            tdm_load_tile_2d(Km + headOff + (size_t)kb * D_MODEL, Ks, 64, 64, D_MODEL);
#else
#pragma unroll
        for (int i = 0; i < 2; ++i) {               // K tile 64x64 straight
            int c = i * 256 + tid;
            int ky = c >> 3, kc = c & 7;
            *(uint4*)(Ks + ky * 72 + kc * 8) =
                *(const uint4*)(Km + headOff + (size_t)(kb + ky) * D_MODEL + kc * 8);
        }
#endif
#pragma unroll
        for (int i = 0; i < 2; ++i) {               // V tile 64x64 transposed scatter
            int c = i * 256 + tid;
            int ky = c >> 3, dc = c & 7;
            uint4 t = *(const uint4*)(V + headOff + (size_t)(kb + ky) * D_MODEL + dc * 8);
            const bf16_t* e = (const bf16_t*)&t;
#pragma unroll
            for (int j = 0; j < 8; ++j) Vt[(dc * 8 + j) * 72 + ky] = e[j];
        }
#if HAVE_TDM
        if (w == 0) __builtin_amdgcn_s_wait_tensorcnt((short)0);
#endif
        __syncthreads();

        v8f s[4];
#pragma unroll
        for (int nt = 0; nt < 4; ++nt) s[nt] = {};
#pragma unroll
        for (int kk = 0; kk < 2; ++kk)
#pragma unroll
            for (int nt = 0; nt < 4; ++nt) {
                v16bf bk = lds_load_b_fragT(Ks + nt * 16 * 72, 72, kk * 32, lane);
                s[nt] = wmma_bf16(aq[kk], bk, s[nt]);
            }
        // causal mask (scale already folded into Q)
#pragma unroll
        for (int nt = 0; nt < 4; ++nt) {
            int col = kb + nt * 16 + ln;
#pragma unroll
            for (int r = 0; r < 8; ++r)
                s[nt][r] = (col > myRowBase + r) ? -3.0e38f : s[nt][r];
        }
        // online softmax (row = 16 lanes of one half-wave + 4 in-lane tiles)
        float nmax[8], alpha[8], psum[8];
#pragma unroll
        for (int r = 0; r < 8; ++r) {
            float m = rmax[r];
#pragma unroll
            for (int nt = 0; nt < 4; ++nt) m = fmaxf(m, s[nt][r]);
#pragma unroll
            for (int off = 8; off >= 1; off >>= 1)
                m = fmaxf(m, __shfl_xor(m, off, 32));
            nmax[r] = m;
            alpha[r] = __expf(rmax[r] - m);
            psum[r] = 0.f;
        }
#pragma unroll
        for (int nt = 0; nt < 4; ++nt)
#pragma unroll
            for (int r = 0; r < 8; ++r) {
                float p = __expf(s[nt][r] - nmax[r]);
                s[nt][r] = p;
                psum[r] += p;
            }
#pragma unroll
        for (int r = 0; r < 8; ++r) {
#pragma unroll
            for (int off = 8; off >= 1; off >>= 1)
                psum[r] += __shfl_xor(psum[r], off, 32);
            rsum[r] = rsum[r] * alpha[r] + psum[r];
            rmax[r] = nmax[r];
        }
#pragma unroll
        for (int nt = 0; nt < 4; ++nt)
#pragma unroll
            for (int r = 0; r < 8; ++r) o[nt][r] *= alpha[r];

        // P (C-layout) -> per-wave LDS -> A-layout fragments
        bf16_t* pw = Ps + w * 16 * 72;
#pragma unroll
        for (int nt = 0; nt < 4; ++nt)
#pragma unroll
            for (int r = 0; r < 8; ++r)
                pw[(r + 8 * hf) * 72 + nt * 16 + ln] = (bf16_t)s[nt][r];
        asm volatile("s_wait_dscnt 0" ::: "memory");  // wave-private region: no barrier needed
        v16bf ap[2];
        ap[0] = lds_load_a_frag(pw, 72, 0, lane);
        ap[1] = lds_load_a_frag(pw, 72, 32, lane);
#pragma unroll
        for (int kk = 0; kk < 2; ++kk)
#pragma unroll
            for (int nt = 0; nt < 4; ++nt) {
                v16bf bv = lds_load_b_fragT(Vt + nt * 16 * 72, 72, kk * 32, lane);
                o[nt] = wmma_bf16(ap[kk], bv, o[nt]);
            }
    }
    // normalize + write context (merged heads layout [B,S,D])
#pragma unroll
    for (int nt = 0; nt < 4; ++nt) {
        int dv = nt * 16 + ln;
#pragma unroll
        for (int r = 0; r < 8; ++r) {
            int row = myRowBase + r;
            ctx[headOff + (size_t)row * D_MODEL + dv] = (bf16_t)(o[nt][r] / rsum[r]);
        }
    }
}

// ---------------------------------------------------------------- launcher
extern "C" void kernel_launch(void* const* d_in, const int* in_sizes, int n_in,
                              void* d_out, int out_size, void* d_ws, size_t ws_size,
                              hipStream_t stream) {
    const float* x  = (const float*)d_in[0];
    // d_in[1] = mask (causal, implicit in kernel)
    const float* Wq = (const float*)d_in[2];
    const float* bq = (const float*)d_in[3];
    const float* Wk = (const float*)d_in[4];
    const float* bk = (const float*)d_in[5];
    const float* Wv = (const float*)d_in[6];
    const float* bv = (const float*)d_in[7];
    const float* Wo = (const float*)d_in[8];
    const float* bo = (const float*)d_in[9];

    const size_t MTOK = (size_t)4 * S_LEN;              // 8192 rows
    const size_t XE   = MTOK * D_MODEL;                 // 8,388,608 elements
    const size_t WE   = (size_t)D_MODEL * D_MODEL;      // 1,048,576 elements

    char* ws = (char*)d_ws;
    size_t off = 0;
    bf16_t* xb  = (bf16_t*)(ws + off); off += XE * 2;
    bf16_t* wqb = (bf16_t*)(ws + off); off += WE * 2;
    bf16_t* wkb = (bf16_t*)(ws + off); off += WE * 2;
    bf16_t* wvb = (bf16_t*)(ws + off); off += WE * 2;
    bf16_t* wob = (bf16_t*)(ws + off); off += WE * 2;
    bf16_t* Qb  = (bf16_t*)(ws + off); off += XE * 2;
    bf16_t* Kb  = (bf16_t*)(ws + off); off += XE * 2;
    bf16_t* Vb  = (bf16_t*)(ws + off); off += XE * 2;
    bf16_t* Cb  = (bf16_t*)(ws + off); off += XE * 2;

    // fp32 -> bf16 staging
    cvt_f32_bf16<<<(int)(XE / 4 / 256), 256, 0, stream>>>(x,  xb,  (int)(XE / 4));
    cvt_f32_bf16<<<(int)(WE / 4 / 256), 256, 0, stream>>>(Wq, wqb, (int)(WE / 4));
    cvt_f32_bf16<<<(int)(WE / 4 / 256), 256, 0, stream>>>(Wk, wkb, (int)(WE / 4));
    cvt_f32_bf16<<<(int)(WE / 4 / 256), 256, 0, stream>>>(Wv, wvb, (int)(WE / 4));
    cvt_f32_bf16<<<(int)(WE / 4 / 256), 256, 0, stream>>>(Wo, wob, (int)(WE / 4));

    dim3 gproj(D_MODEL / 128, (int)(MTOK / 128));       // 8 x 64
    // Q projection carries the 1/sqrt(DK)=0.125 softmax scale.
    gemm_bf16<false><<<gproj, 256, 0, stream>>>(xb, wqb, bq, Qb, (int)MTOK, D_MODEL, D_MODEL, 0.125f);
    gemm_bf16<false><<<gproj, 256, 0, stream>>>(xb, wkb, bk, Kb, (int)MTOK, D_MODEL, D_MODEL, 1.0f);
    gemm_bf16<false><<<gproj, 256, 0, stream>>>(xb, wvb, bv, Vb, (int)MTOK, D_MODEL, D_MODEL, 1.0f);

    dim3 gattn(S_LEN / 128, 4 * N_HEADS);               // 16 x 64
    flash_attn<<<gattn, 256, 0, stream>>>(Qb, Kb, Vb, Cb);

    gemm_bf16<true><<<gproj, 256, 0, stream>>>(Cb, wob, bo, d_out, (int)MTOK, D_MODEL, D_MODEL, 1.0f);

    (void)in_sizes; (void)n_in; (void)out_size; (void)ws_size;
}
